// PixelAttentionBlock__48868137894300
// MI455X (gfx1250) — matrix operand order repaired and checked by reading, not compiled, not run on hardware
//
#include <hip/hip_runtime.h>

// ---------------------------------------------------------------------------
// PixelAttentionBlock on MI455X (gfx1250), wave32 + WMMA + TDM.
//   kf   = relu(BN(wk @ x))                   -> bf16 WMMA, fused BN+ReLU
//   sim  = softmax(kf^T kf * KC^-0.5)         -> bf16 WMMA, TDM-staged B panel
//   v    = relu(conv3x3(x)+b1); relu(wv2@v+b2)-> bf16 WMMA (9-tap implicit GEMM)
//   ctx  = sim @ v^T                          -> bf16 WMMA, double-buffered TDM
//                                                pipeline (f32 WMMA fallback)
// ---------------------------------------------------------------------------

typedef __attribute__((ext_vector_type(16))) __bf16 v16bf;
typedef __attribute__((ext_vector_type(8)))  __bf16 v8bf;
typedef __attribute__((ext_vector_type(8)))  float  v8f;
typedef __attribute__((ext_vector_type(2)))  float  v2f;
typedef __attribute__((ext_vector_type(4)))  unsigned int u32x4;
typedef __attribute__((ext_vector_type(8)))  int i32x8;
typedef __attribute__((ext_vector_type(4)))  int i32x4;

static constexpr int Bn  = 4;
static constexpr int Cc  = 512;
static constexpr int Hh  = 64;
static constexpr int Ww  = 80;
static constexpr int Nn  = Hh * Ww;   // 5120
static constexpr int KCc = 256;
static constexpr int VCc = 512;
static constexpr float kEps   = 1e-5f;
static constexpr float kScale = 0.0625f; // KC^-0.5

#if __has_builtin(__builtin_amdgcn_tensor_load_to_lds)
#define ENABLE_TDM 1
#else
#define ENABLE_TDM 0
#endif

// ---------------- fragment loaders (contiguous K order, same for A and B) ----------------

// A: 16x32 bf16 tile, row-major, row stride lda.
// lane<16: K = 0..15; lane>=16: K = 16..31; M = lane&15.
static __device__ __forceinline__ v16bf frag_a_bf16(const __bf16* __restrict__ tile,
                                                    int lda, int lane) {
  const int m  = lane & 15;
  const int kb = (lane >> 4) << 4;          // 0 or 16
  const __bf16* r = tile + (size_t)m * lda;
  return *(const v16bf*)(r + kb);
}

// B: 32x16 tile stored transposed (per-column K-contiguous rows, stride ldb).
// lane<16 holds K=0..15 of column n=lane; lane>=16 holds K=16..31.
static __device__ __forceinline__ v16bf frag_b_bf16(const __bf16* bt,
                                                    int ldb, int lane) {
  const int n  = lane & 15;
  const int kh = (lane >> 4) << 4;          // 0 or 16
  return *(const v16bf*)(bt + (size_t)n * ldb + kh);
}

static __device__ __forceinline__ v16bf zero_v16bf() {
  v16bf f;
#pragma unroll
  for (int i = 0; i < 16; ++i) f[i] = (__bf16)0.0f;
  return f;
}

// --------------------- TDM / LDS panel staging -----------------------------
// 2-D tile (tileRows x tileCols bf16, row pitch ld elements), global -> LDS.
// TDM path: wave 0 issues one tensor_load_to_lds (EXEC-independent, tracked
// by TENSORcnt). panel_issue / panel_wait are split so callers can overlap
// the DMA with WMMA compute (double buffering).

#if ENABLE_TDM
static __device__ __forceinline__ void tdm_load_2d(const void* sdst, const void* gsrc,
                                                   unsigned tileRows, unsigned tileCols,
                                                   unsigned ld) {
  const unsigned lds_off = (unsigned)(size_t)sdst;
  const unsigned long long ga = (unsigned long long)gsrc;
  u32x4 g0;
  g0[0] = 1u;                                             // count=1 (user mode)
  g0[1] = lds_off;                                        // lds_addr
  g0[2] = (unsigned)ga;                                   // global_addr[31:0]
  g0[3] = (unsigned)((ga >> 32) & 0x01FFFFFFull) | (2u << 30); // addr[56:32]|type=2
  i32x8 g1;
  g1[0] = (int)(1u << 16);                                // data_size=1 -> 2 bytes
  g1[1] = (int)((tileCols & 0xFFFFu) << 16);              // tensor_dim0 lo
  g1[2] = (int)(((tileCols >> 16) & 0xFFFFu) |            // tensor_dim0 hi
                ((tileRows & 0xFFFFu) << 16));            // tensor_dim1 lo
  g1[3] = (int)(((tileRows >> 16) & 0xFFFFu) |            // tensor_dim1 hi
                ((tileCols & 0xFFFFu) << 16));            // tile_dim0
  g1[4] = (int)(tileRows & 0xFFFFu);                      // tile_dim1 (tile_dim2=0)
  g1[5] = (int)ld;                                        // tensor_dim0_stride lo
  g1[6] = 0;
  g1[7] = 0;
  i32x4 zg = {};
#if __clang_major__ >= 23
  i32x8 zg8 = {};
  __builtin_amdgcn_tensor_load_to_lds(g0, g1, zg, zg, zg8, 0);
#else
  __builtin_amdgcn_tensor_load_to_lds(g0, g1, zg, zg, 0);
#endif
}
#endif

static __device__ __forceinline__ void panel_issue(__bf16* sB, const __bf16* src,
                                                   int rows, int cols, int ld) {
#if ENABLE_TDM
  if (threadIdx.x < 32)
    tdm_load_2d((const void*)sB, (const void*)src,
                (unsigned)rows, (unsigned)cols, (unsigned)ld);
#else
  const int vecs = cols >> 3;
  for (int idx = threadIdx.x; idx < rows * vecs; idx += 256) {
    const int r = idx / vecs, c = (idx - r * vecs) << 3;
    *(v8bf*)(sB + r * cols + c) = *(const v8bf*)(src + (size_t)r * ld + c);
  }
#endif
}

static __device__ __forceinline__ void panel_wait() {
#if ENABLE_TDM
  if (threadIdx.x < 32) __builtin_amdgcn_s_wait_tensorcnt(0);
#endif
  __syncthreads();
}

static __device__ __forceinline__ void stage_panel(__bf16* sB, const __bf16* src,
                                                   int rows, int cols, int ld) {
  panel_issue(sB, src, rows, cols, ld);
  panel_wait();
}

// ---------------------------- pack / convert -------------------------------

__global__ void k_cvt_bf16(const float* __restrict__ src, __bf16* __restrict__ dst,
                           size_t n) {
  size_t i = (size_t)blockIdx.x * 256 + threadIdx.x;
  if (i < n) dst[i] = (__bf16)src[i];
}

// xT[b][n][c] = x[b][c][n]
__global__ void k_pack_xT(const float* __restrict__ x, __bf16* __restrict__ xT) {
  size_t i = (size_t)blockIdx.x * 256 + threadIdx.x;
  const size_t total = (size_t)Bn * Nn * Cc;
  if (i >= total) return;
  int c = (int)(i % Cc);
  size_t t = i / Cc;
  int n = (int)(t % Nn);
  int b = (int)(t / Nn);
  xT[i] = (__bf16)x[((size_t)b * Cc + c) * Nn + n];
}

// wv1p[(tap*VC + o)*C + ci] = wv1[o][ci][ty][tx], tap = ty*3+tx
__global__ void k_pack_wv1(const float* __restrict__ wv1, __bf16* __restrict__ wv1p) {
  size_t i = (size_t)blockIdx.x * 256 + threadIdx.x;
  const size_t total = (size_t)9 * VCc * Cc;
  if (i >= total) return;
  int ci  = (int)(i % Cc);
  size_t t = i / Cc;
  int o   = (int)(t % VCc);
  int tap = (int)(t / VCc);
  wv1p[i] = (__bf16)wv1[((size_t)o * Cc + ci) * 9 + tap];
}

// ------------------------- GEMM 1: projection ------------------------------

__global__ __launch_bounds__(256) void k_gemm_kf(
    const __bf16* __restrict__ wkb, const __bf16* __restrict__ xT,
    const float* __restrict__ bk, const float* __restrict__ gamma,
    const float* __restrict__ beta, const float* __restrict__ rmean,
    const float* __restrict__ rvar, __bf16* __restrict__ kfT) {
  const int lane = threadIdx.x & 31, wave = threadIdx.x >> 5;
  const int n0 = (blockIdx.x * 8 + wave) * 16;
  const int m0 = blockIdx.y * 16;
  const int b  = blockIdx.z;
  const __bf16* A  = wkb + (size_t)m0 * Cc;
  const __bf16* Bt = xT + ((size_t)b * Nn + n0) * Cc;
  v8f acc = {};
  for (int k = 0; k < Cc; k += 32) {
    v16bf a  = frag_a_bf16(A + k, Cc, lane);
    v16bf bb = frag_b_bf16(Bt + k, Cc, lane);
    acc = __builtin_amdgcn_wmma_f32_16x16x32_bf16(false, a, false, bb,
                                                  (short)0, acc, false, false);
  }
  const int nn = lane & 15, mb = (lane >> 4) << 3;
  v8bf outv;
#pragma unroll
  for (int i = 0; i < 8; ++i) {
    const int ko = m0 + mb + i;
    float v = acc[i] + bk[ko];
    v = (v - rmean[ko]) * (gamma[ko] * rsqrtf(rvar[ko] + kEps)) + beta[ko];
    outv[i] = (__bf16)fmaxf(v, 0.0f);
  }
  *(v8bf*)(kfT + ((size_t)b * Nn + n0 + nn) * KCc + m0 + mb) = outv;
}

// ------------------------- GEMM 2: similarity ------------------------------
// Macro-tile 128(M) x 128(N) per block; B panel (128 x KC bf16 = 64KB) staged
// into LDS by the TDM; each wave computes a 16x128 strip (8 WMMA per A frag).

__global__ __launch_bounds__(256) void k_gemm_sim(
    const __bf16* __restrict__ kfT, float* __restrict__ sim) {
  __shared__ __bf16 sB[128 * KCc];
  const int lane = threadIdx.x & 31, wave = threadIdx.x >> 5;
  const int n0 = blockIdx.x * 128;
  const int m0 = blockIdx.y * 128;
  const int b  = blockIdx.z;
  stage_panel(sB, kfT + ((size_t)b * Nn + n0) * KCc, 128, KCc, KCc);

  const __bf16* A = kfT + ((size_t)b * Nn + m0 + wave * 16) * KCc;
  v8f acc[8] = {};
  for (int k = 0; k < KCc; k += 32) {
    v16bf a = frag_a_bf16(A + k, KCc, lane);
#pragma unroll
    for (int t = 0; t < 8; ++t) {
      v16bf bb = frag_b_bf16(sB + (size_t)(t * 16) * KCc + k, KCc, lane);
      acc[t] = __builtin_amdgcn_wmma_f32_16x16x32_bf16(false, a, false, bb,
                                                       (short)0, acc[t], false, false);
    }
  }
  const int nn = lane & 15, mb = (lane >> 4) << 3;
  const int mrow = m0 + wave * 16 + mb;
#pragma unroll
  for (int t = 0; t < 8; ++t) {
    float* dst = sim + ((size_t)b * Nn + mrow) * Nn + n0 + t * 16 + nn;
#pragma unroll
    for (int i = 0; i < 8; ++i) dst[(size_t)i * Nn] = acc[t][i] * kScale;
  }
}

// ------------------------------ softmax ------------------------------------

__global__ __launch_bounds__(256) void k_softmax(float* __restrict__ sim,
                                                 __bf16* __restrict__ simbf) {
  __shared__ float red[8];
  const size_t row = blockIdx.x;
  float* r = sim + row * Nn;
  const int tid = threadIdx.x;
  float mx = -3.4e38f;
  for (int i = tid; i < Nn; i += 256) mx = fmaxf(mx, r[i]);
#pragma unroll
  for (int o = 16; o; o >>= 1) mx = fmaxf(mx, __shfl_xor(mx, o));
  if ((tid & 31) == 0) red[tid >> 5] = mx;
  __syncthreads();
  float m = red[0];
#pragma unroll
  for (int i = 1; i < 8; ++i) m = fmaxf(m, red[i]);
  __syncthreads();
  float s = 0.0f;
  for (int i = tid; i < Nn; i += 256) {
    float e = __expf(r[i] - m);
    r[i] = e;
    s += e;
  }
#pragma unroll
  for (int o = 16; o; o >>= 1) s += __shfl_xor(s, o);
  if ((tid & 31) == 0) red[tid >> 5] = s;
  __syncthreads();
  float tot = 0.0f;
#pragma unroll
  for (int i = 0; i < 8; ++i) tot += red[i];
  const float inv = 1.0f / tot;
  if (simbf) {
    __bf16* rb = simbf + row * Nn;
    for (int i = tid; i < Nn; i += 256) {
      float v = r[i] * inv;
      r[i] = v;
      rb[i] = (__bf16)v;
    }
  } else {
    for (int i = tid; i < Nn; i += 256) r[i] *= inv;
  }
}

// -------------------- 3x3 conv as 9-tap implicit GEMM ----------------------

__global__ __launch_bounds__(256) void k_conv(
    const __bf16* __restrict__ wv1p, const __bf16* __restrict__ xT,
    const float* __restrict__ bv1, __bf16* __restrict__ v1t) {
  const int lane = threadIdx.x & 31, wave = threadIdx.x >> 5;
  const int n0 = (blockIdx.x * 8 + wave) * 16;
  const int m0 = blockIdx.y * 16;
  const int b  = blockIdx.z;
  const int nn = lane & 15;
  const int n  = n0 + nn;
  const int h  = n / Ww, w = n % Ww;
  const int kh = (lane >> 4) << 4;
  v8f acc = {};
#pragma unroll
  for (int tap = 0; tap < 9; ++tap) {
    const int dy = tap / 3 - 1, dx = tap % 3 - 1;
    const int hh = h + dy, ww = w + dx;
    const bool valid = (hh >= 0) & (hh < Hh) & (ww >= 0) & (ww < Ww);
    const __bf16* A    = wv1p + ((size_t)tap * VCc + m0) * Cc;
    const __bf16* brow = xT + ((size_t)b * Nn + hh * Ww + ww) * Cc + kh;
    for (int k = 0; k < Cc; k += 32) {
      v16bf a  = frag_a_bf16(A + k, Cc, lane);
      v16bf bb = valid ? *(const v16bf*)(brow + k) : zero_v16bf();
      acc = __builtin_amdgcn_wmma_f32_16x16x32_bf16(false, a, false, bb,
                                                    (short)0, acc, false, false);
    }
  }
  const int mb = (lane >> 4) << 3;
  v8bf outv;
#pragma unroll
  for (int i = 0; i < 8; ++i)
    outv[i] = (__bf16)fmaxf(acc[i] + bv1[m0 + mb + i], 0.0f);
  *(v8bf*)(v1t + ((size_t)b * Nn + n) * VCc + m0 + mb) = outv;
}

// ------------------------- GEMM 3: value 1x1 -------------------------------

__global__ __launch_bounds__(256) void k_gemm_v2(
    const __bf16* __restrict__ wv2b, const __bf16* __restrict__ v1t,
    const float* __restrict__ bv2, float* __restrict__ v2,
    __bf16* __restrict__ v2bf) {
  const int lane = threadIdx.x & 31, wave = threadIdx.x >> 5;
  const int n0 = (blockIdx.x * 8 + wave) * 16;
  const int m0 = blockIdx.y * 16;
  const int b  = blockIdx.z;
  const __bf16* A  = wv2b + (size_t)m0 * VCc;
  const __bf16* Bt = v1t + ((size_t)b * Nn + n0) * VCc;
  v8f acc = {};
  for (int k = 0; k < VCc; k += 32) {
    v16bf a  = frag_a_bf16(A + k, VCc, lane);
    v16bf bb = frag_b_bf16(Bt + k, VCc, lane);
    acc = __builtin_amdgcn_wmma_f32_16x16x32_bf16(false, a, false, bb,
                                                  (short)0, acc, false, false);
  }
  const int nn = lane & 15, mb = (lane >> 4) << 3;
#pragma unroll
  for (int i = 0; i < 8; ++i) {
    const int c = m0 + mb + i;
    const float v = fmaxf(acc[i] + bv2[c], 0.0f);
    const size_t o = ((size_t)b * VCc + c) * Nn + n0 + nn;
    v2[o] = v;
    v2bf[o] = (__bf16)v;
  }
}

// -------------------- GEMM 4a: context, bf16 fast path ---------------------
// ctx[b][c][n] = sum_m v2[c][m] * sim[n][m]. Macro 128x128; sim K-chunks of
// 256 double-buffered through LDS: the TDM for chunk i+1 is issued before the
// WMMAs of chunk i, overlapped via TENSORcnt + split barriers.

__global__ __launch_bounds__(256) void k_context_bf16(
    const __bf16* __restrict__ v2bf, const __bf16* __restrict__ simbf,
    float* __restrict__ ctx) {
  __shared__ __bf16 sB[2][128 * 256];    // 2 x 64KB panels
  const int lane = threadIdx.x & 31, wave = threadIdx.x >> 5;
  const int n0 = blockIdx.x * 128;
  const int m0 = blockIdx.y * 128;
  const int b  = blockIdx.z;
  const __bf16* A    = v2bf + ((size_t)b * VCc + m0 + wave * 16) * Nn;
  const __bf16* base = simbf + ((size_t)b * Nn + n0) * Nn;
  constexpr int nchunks = Nn / 256;      // 20

  panel_issue(sB[0], base, 128, 256, Nn);
  panel_wait();

  v8f acc[8] = {};
  for (int ic = 0; ic < nchunks; ++ic) {
    const int kc = ic * 256;
    if (ic + 1 < nchunks)                // prefetch next chunk into other panel
      panel_issue(sB[(ic + 1) & 1], base + kc + 256, 128, 256, Nn);
    const __bf16* cur = sB[ic & 1];
    for (int k = 0; k < 256; k += 32) {
      v16bf a = frag_a_bf16(A + kc + k, Nn, lane);
#pragma unroll
      for (int t = 0; t < 8; ++t) {
        v16bf bb = frag_b_bf16(cur + (size_t)(t * 16) * 256 + k, 256, lane);
        acc[t] = __builtin_amdgcn_wmma_f32_16x16x32_bf16(false, a, false, bb,
                                                         (short)0, acc[t], false, false);
      }
    }
    __syncthreads();                     // all waves done reading 'cur'
    if (ic + 1 < nchunks) panel_wait();  // publish next panel
  }

  const int nn = lane & 15, mb = (lane >> 4) << 3;
  const int crow = m0 + wave * 16 + mb;
#pragma unroll
  for (int t = 0; t < 8; ++t) {
    float* dst = ctx + ((size_t)b * VCc + crow) * Nn + n0 + t * 16 + nn;
#pragma unroll
    for (int i = 0; i < 8; ++i) dst[(size_t)i * Nn] = acc[t][i];
  }
}

// -------------------- GEMM 4b: context, f32 fallback -----------------------

__global__ __launch_bounds__(256) void k_context_f32(
    const float* __restrict__ v2, const float* __restrict__ sim,
    float* __restrict__ ctx) {
  const int lane = threadIdx.x & 31, wave = threadIdx.x >> 5;
  const int n0 = (blockIdx.x * 8 + wave) * 16;
  const int m0 = blockIdx.y * 16;
  const int b  = blockIdx.z;
  const int nn = lane & 15;
  const int kb = (lane >> 4) << 1;
  const float* Arow = v2  + ((size_t)b * VCc + m0 + (lane & 15)) * Nn;
  const float* Brow = sim + ((size_t)b * Nn + n0 + nn) * Nn;
  v8f acc = {};
#pragma unroll 8
  for (int k = 0; k < Nn; k += 4) {
    v2f a  = *(const v2f*)(Arow + k + kb);
    v2f bb = *(const v2f*)(Brow + k + kb);
    acc = __builtin_amdgcn_wmma_f32_16x16x4_f32(false, a, false, bb,
                                                (short)0, acc, false, false);
  }
  const int mb = (lane >> 4) << 3;
#pragma unroll
  for (int i = 0; i < 8; ++i)
    ctx[((size_t)b * VCc + m0 + mb + i) * Nn + n0 + nn] = acc[i];
}

// ------------------------------- launcher ----------------------------------

extern "C" void kernel_launch(void* const* d_in, const int* in_sizes, int n_in,
                              void* d_out, int out_size, void* d_ws, size_t ws_size,
                              hipStream_t stream) {
  (void)in_sizes; (void)n_in; (void)out_size;
  const float* x     = (const float*)d_in[0];
  const float* wk    = (const float*)d_in[1];
  const float* bk    = (const float*)d_in[2];
  const float* gamma = (const float*)d_in[3];
  const float* beta  = (const float*)d_in[4];
  const float* rmean = (const float*)d_in[5];
  const float* rvar  = (const float*)d_in[6];
  const float* wv1   = (const float*)d_in[7];
  const float* bv1   = (const float*)d_in[8];
  const float* wv2   = (const float*)d_in[9];
  const float* bv2   = (const float*)d_in[10];

  float* ctx = (float*)d_out;
  float* sim = ctx + (size_t)Bn * VCc * Nn;   // outputs: (context, sim) flat

  char* p = (char*)d_ws;
  auto take = [&](size_t bytes) -> char* {
    char* r = p;
    p += (bytes + 255) & ~(size_t)255;
    return r;
  };
  __bf16* xT   = (__bf16*)take((size_t)Bn * Nn * Cc * 2);
  __bf16* wkb  = (__bf16*)take((size_t)KCc * Cc * 2);
  __bf16* kfT  = (__bf16*)take((size_t)Bn * Nn * KCc * 2);
  __bf16* wv1p = (__bf16*)take((size_t)9 * VCc * Cc * 2);
  __bf16* wv2b = (__bf16*)take((size_t)VCc * VCc * 2);
  __bf16* v1t  = (__bf16*)take((size_t)Bn * Nn * VCc * 2);
  float*  v2   = (float*)take((size_t)Bn * VCc * Nn * 4);
  __bf16* v2bf = (__bf16*)take((size_t)Bn * VCc * Nn * 2);
  // bf16 sim mirror (210 MB) only if the workspace is big enough.
  const size_t simbf_bytes = (size_t)Bn * Nn * Nn * 2;
  __bf16* simbf = nullptr;
  if ((size_t)(p - (char*)d_ws) + simbf_bytes <= ws_size)
    simbf = (__bf16*)take(simbf_bytes);

  // pack / convert
  {
    size_t n1 = (size_t)KCc * Cc;
    k_cvt_bf16<<<(unsigned)((n1 + 255) / 256), 256, 0, stream>>>(wk, wkb, n1);
    size_t n2 = (size_t)VCc * VCc;
    k_cvt_bf16<<<(unsigned)((n2 + 255) / 256), 256, 0, stream>>>(wv2, wv2b, n2);
    size_t n3 = (size_t)9 * VCc * Cc;
    k_pack_wv1<<<(unsigned)((n3 + 255) / 256), 256, 0, stream>>>(wv1, wv1p);
    size_t n4 = (size_t)Bn * Nn * Cc;
    k_pack_xT<<<(unsigned)((n4 + 255) / 256), 256, 0, stream>>>(x, xT);
  }

  // attention path
  dim3 gKf(Nn / 128, KCc / 16, Bn);
  k_gemm_kf<<<gKf, 256, 0, stream>>>(wkb, xT, bk, gamma, beta, rmean, rvar, kfT);
  dim3 gSim(Nn / 128, Nn / 128, Bn);
  k_gemm_sim<<<gSim, 256, 0, stream>>>(kfT, sim);
  k_softmax<<<Bn * Nn, 256, 0, stream>>>(sim, simbf);

  // value path
  dim3 gV(Nn / 128, VCc / 16, Bn);
  k_conv<<<gV, 256, 0, stream>>>(wv1p, xT, bv1, v1t);
  k_gemm_v2<<<gV, 256, 0, stream>>>(wv2b, v1t, bv2, v2, v2bf);

  // aggregation
  if (simbf) {
    dim3 gCtx(Nn / 128, VCc / 128, Bn);
    k_context_bf16<<<gCtx, 256, 0, stream>>>(v2bf, simbf, ctx);
  } else {
    k_context_f32<<<gV, 256, 0, stream>>>(v2, sim, ctx);
  }
}
